// Model_69140383531671
// MI455X (gfx1250) — compile-verified, matrix operand order checked by reading, not run on hardware
//
#include <hip/hip_runtime.h>

typedef __attribute__((ext_vector_type(16))) _Float16 v16h;
typedef __attribute__((ext_vector_type(8)))  _Float16 v8h;
typedef __attribute__((ext_vector_type(2)))  _Float16 h2;
typedef __attribute__((ext_vector_type(8)))  float    v8f;

#define WAVES 4
#define TPB   (WAVES * 32)

#define WMMA_F16(a, b, c) \
  __builtin_amdgcn_wmma_f32_16x16x32_f16(false, (a), false, (b), (short)0, (c), false, false)
#define WAIT_DS() asm volatile("s_wait_dscnt 0" ::: "memory")

// CDNA5 16-bit A/B fragment K mapping (CDNA5 ISA, 16-bit A 16x32):
// element j on lane l covers K = 16*(j>=8) + 8*(l>=16) + (j&7); row/col = l&15.
__device__ __forceinline__ int kmap(int lane, int j) {
  return ((j & 8) << 1) | ((lane & 16) >> 1) | (j & 7);
}

// Build B fragment (KxN tile, N-tile index ntile) from a row-major [rows x cols] f32 weight.
__device__ __forceinline__ v16h load_bfrag(const float* __restrict__ W, int rows, int cols,
                                           int ntile, int lane) {
  v16h b;
  int n = ntile * 16 + (lane & 15);
  #pragma unroll
  for (int j = 0; j < 16; ++j) {
    int k = kmap(lane, j);
    float v = (k < rows && n < cols) ? W[k * cols + n] : 0.0f;
    b[j] = (_Float16)v;
  }
  return b;
}

// Build A fragment from a wave-private LDS tile laid out [16 rows][32 K] f16.
// Two contiguous 16B runs per lane -> 2x ds_load_b128.
__device__ __forceinline__ v16h load_afrag_lds(const _Float16* t, int lane) {
  int m  = lane & 15;
  int kb = (lane & 16) >> 1;                       // 0 or 8
  const v8h* p = (const v8h*)(t + m * 32 + kb);    // 16B aligned
  v8h lo = p[0];
  v8h hi = p[2];                                   // +16 halves
  v16h a;
  #pragma unroll
  for (int j = 0; j < 8; ++j) { a[j] = lo[j]; a[8 + j] = hi[j]; }
  return a;
}

// ---------------- Encoder: per-edge MLP (12->20->20->20->150) + msg + scatter ----------------
__global__ __launch_bounds__(TPB) void enc_edge_kernel(
    const float* __restrict__ x, const int* __restrict__ ei, const float* __restrict__ ea,
    const float* __restrict__ ew1, const float* __restrict__ eb1,
    const float* __restrict__ ew2, const float* __restrict__ eb2,
    const float* __restrict__ ew3, const float* __restrict__ eb3,
    const float* __restrict__ ew4, const float* __restrict__ eb4,
    float* __restrict__ s_enc, float* __restrict__ cnt, int N, int E)
{
  __shared__ _Float16 sh_in [WAVES][16 * 32];
  __shared__ _Float16 sh_act[WAVES][16 * 32];
  __shared__ float    sh_out[WAVES][16 * 160];
  __shared__ float    sh_xs [WAVES][64];
  __shared__ int      sh_dst[WAVES][16];

  const int lane = threadIdx.x & 31;
  const int wid  = threadIdx.x >> 5;
  _Float16* in16 = sh_in[wid];
  _Float16* act  = sh_act[wid];
  float*    wout = sh_out[wid];
  float*    xs   = sh_xs[wid];
  int*      dsti = sh_dst[wid];

  const int n15   = lane & 15;
  const int mbase = (lane & 16) ? 8 : 0;

  // zero input tile once (pad cols 12..31 stay zero forever)
  {
    unsigned* z = (unsigned*)in16;   // 256 dwords
    #pragma unroll
    for (int i = 0; i < 8; ++i) z[lane + 32 * i] = 0u;
  }

  v16h b1[2], b2[2], b3[2], b4[10];
  float bi1[2], bi2[2], bi3[2], bi4[10];
  #pragma unroll
  for (int t = 0; t < 2; ++t) {
    b1[t] = load_bfrag(ew1, 12, 20, t, lane);
    b2[t] = load_bfrag(ew2, 20, 20, t, lane);
    b3[t] = load_bfrag(ew3, 20, 20, t, lane);
    int col = t * 16 + n15;
    bi1[t] = (col < 20) ? eb1[col] : 0.0f;
    bi2[t] = (col < 20) ? eb2[col] : 0.0f;
    bi3[t] = (col < 20) ? eb3[col] : 0.0f;
  }
  #pragma unroll
  for (int t = 0; t < 10; ++t) {
    b4[t] = load_bfrag(ew4, 20, 150, t, lane);
    int col = t * 16 + n15;
    bi4[t] = (col < 150) ? eb4[col] : 0.0f;
  }
  WAIT_DS();

  const int nTiles = (E + 15) >> 4;
  const int gw = blockIdx.x * WAVES + wid;
  const int nw = gridDim.x * WAVES;

  for (int tile = gw; tile < nTiles; tile += nw) {
    const int e0 = tile << 4;

    // Stage the 16x12 contiguous edge_attr tile into LDS f16 (coalesced b64 loads).
    {
      int row  = lane >> 1;
      int colb = (lane & 1) * 6;
      float a0 = 0.f, a1 = 0.f, a2 = 0.f, a3 = 0.f, a4 = 0.f, a5 = 0.f;
      if (e0 + row < E) {
        const float* p = ea + (size_t)(e0 + row) * 12 + colb;   // 8B aligned
        float2 u0 = *(const float2*)(p + 0);
        float2 u1 = *(const float2*)(p + 2);
        float2 u2 = *(const float2*)(p + 4);
        a0 = u0.x; a1 = u0.y; a2 = u1.x; a3 = u1.y; a4 = u2.x; a5 = u2.y;
      }
      h2* q = (h2*)(in16 + row * 32 + colb);                    // 4B aligned
      h2 t0; t0[0] = (_Float16)a0; t0[1] = (_Float16)a1;
      h2 t1; t1[0] = (_Float16)a2; t1[1] = (_Float16)a3;
      h2 t2; t2[0] = (_Float16)a4; t2[1] = (_Float16)a5;
      q[0] = t0; q[1] = t1; q[2] = t2;
    }
    WAIT_DS();
    v16h a = load_afrag_lds(in16, lane);

    // Layer 1 (pairwise WMMA issue to cover WMMA->VALU hazard)
    {
      v8f c0 = {}, c1 = {};
      c0 = WMMA_F16(a, b1[0], c0);
      c1 = WMMA_F16(a, b1[1], c1);
      #pragma unroll
      for (int r = 0; r < 8; ++r) {
        float v0 = c0[r] + bi1[0];
        float v1 = c1[r] + bi1[1];
        act[(mbase + r) * 32 + n15]      = (_Float16)(v0 > 0.0f ? v0 : 0.0f);
        act[(mbase + r) * 32 + 16 + n15] = (_Float16)(v1 > 0.0f ? v1 : 0.0f);
      }
    }
    WAIT_DS();
    a = load_afrag_lds(act, lane);

    // Layer 2
    {
      v8f c0 = {}, c1 = {};
      c0 = WMMA_F16(a, b2[0], c0);
      c1 = WMMA_F16(a, b2[1], c1);
      #pragma unroll
      for (int r = 0; r < 8; ++r) {
        float v0 = c0[r] + bi2[0];
        float v1 = c1[r] + bi2[1];
        act[(mbase + r) * 32 + n15]      = (_Float16)(v0 > 0.0f ? v0 : 0.0f);
        act[(mbase + r) * 32 + 16 + n15] = (_Float16)(v1 > 0.0f ? v1 : 0.0f);
      }
    }
    WAIT_DS();
    a = load_afrag_lds(act, lane);

    // Layer 3
    {
      v8f c0 = {}, c1 = {};
      c0 = WMMA_F16(a, b3[0], c0);
      c1 = WMMA_F16(a, b3[1], c1);
      #pragma unroll
      for (int r = 0; r < 8; ++r) {
        float v0 = c0[r] + bi3[0];
        float v1 = c1[r] + bi3[1];
        act[(mbase + r) * 32 + n15]      = (_Float16)(v0 > 0.0f ? v0 : 0.0f);
        act[(mbase + r) * 32 + 16 + n15] = (_Float16)(v1 > 0.0f ? v1 : 0.0f);
      }
    }
    WAIT_DS();
    a = load_afrag_lds(act, lane);

    // Layer 4: [16x20] x [20x150] -> [16x150] f32 in LDS
    #pragma unroll
    for (int t = 0; t < 10; t += 2) {
      v8f c0 = {}, c1 = {};
      c0 = WMMA_F16(a, b4[t], c0);
      c1 = WMMA_F16(a, b4[t + 1], c1);
      #pragma unroll
      for (int r = 0; r < 8; ++r) {
        wout[(mbase + r) * 160 + t * 16 + n15]      = c0[r] + bi4[t];
        wout[(mbase + r) * 160 + t * 16 + 16 + n15] = c1[r] + bi4[t + 1];
      }
    }

    // Gather src coords + dst, bump counts
    if (lane < 16) {
      int ee = e0 + lane;
      if (ee < E) {
        int s = ei[ee];
        int d = ei[E + ee];
        xs[lane * 4 + 0] = x[s * 3 + 0];
        xs[lane * 4 + 1] = x[s * 3 + 1];
        xs[lane * 4 + 2] = x[s * 3 + 2];
        dsti[lane] = d;
        atomicAdd(&cnt[d], 1.0f);
      } else {
        dsti[lane] = -1;
      }
    }
    WAIT_DS();

    // msg[e][o] = sum_i x[src_e][i] * W[e][i*50+o]; scatter-add into s_enc[dst][o]
    for (int t = 0; t < 25; ++t) {
      int idx = lane + 32 * t;  // 0..799
      int el  = idx / 50;
      int o   = idx - el * 50;
      int d   = dsti[el];
      if (d >= 0) {
        const float* wr = wout + el * 160;
        float m = xs[el * 4 + 0] * wr[o]
                + xs[el * 4 + 1] * wr[50 + o]
                + xs[el * 4 + 2] * wr[100 + o];
        atomicAdd(&s_enc[d * 50 + o], m);
      }
    }
  }
}

// ---------------- Finalize h and accumulate pooled sum ----------------
__global__ __launch_bounds__(256) void h_pool_kernel(
    const float* __restrict__ s_enc, const float* __restrict__ cnt,
    const float* __restrict__ x, const float* __restrict__ enc_root,
    const float* __restrict__ enc_bias, float* __restrict__ pooled, int N)
{
  __shared__ float sh[50];
  if (threadIdx.x < 50) sh[threadIdx.x] = 0.0f;
  __syncthreads();
  int idx = blockIdx.x * blockDim.x + threadIdx.x;
  if (idx < N * 50) {
    int n = idx / 50;
    int o = idx - n * 50;
    float c = cnt[n]; c = c < 1.0f ? 1.0f : c;
    float h = s_enc[idx] / c
            + x[n * 3 + 0] * enc_root[o]
            + x[n * 3 + 1] * enc_root[50 + o]
            + x[n * 3 + 2] * enc_root[100 + o]
            + enc_bias[o];
    atomicAdd(&sh[o], h);
  }
  __syncthreads();
  if (threadIdx.x < 50) atomicAdd(&pooled[threadIdx.x], sh[threadIdx.x]);
}

// ---------------- Latent MLP (single block) + fold decoder constants ----------------
__global__ __launch_bounds__(64) void latent_kernel(
    const float* __restrict__ pooled, int N,
    const float* __restrict__ fw1, const float* __restrict__ fb1,
    const float* __restrict__ g1, const float* __restrict__ be1,
    const float* __restrict__ rm1, const float* __restrict__ rv1,
    const float* __restrict__ fw2, const float* __restrict__ fb2,
    const float* __restrict__ g2, const float* __restrict__ be2,
    const float* __restrict__ rm2, const float* __restrict__ rv2,
    const float* __restrict__ fw3, const float* __restrict__ fb3,
    const float* __restrict__ dw1, const float* __restrict__ db1,
    const float* __restrict__ dec_root, const float* __restrict__ dec_bias,
    float* __restrict__ latent_ws, float* __restrict__ bias1_eff,
    float* __restrict__ c3, float* __restrict__ out_latent)
{
  __shared__ float p[50], f1[50], f2[50], lat[50];
  int tid = threadIdx.x;
  if (tid < 50) p[tid] = pooled[tid] / (float)N;
  __syncthreads();
  if (tid < 50) {
    float acc = fb1[tid];
    for (int i = 0; i < 50; ++i) acc += p[i] * fw1[i * 50 + tid];
    acc = (acc - rm1[tid]) * (g1[tid] * rsqrtf(rv1[tid] + 1e-5f)) + be1[tid];
    f1[tid] = acc > 0.0f ? acc : 0.0f;
  }
  __syncthreads();
  if (tid < 50) {
    float acc = fb2[tid];
    for (int i = 0; i < 50; ++i) acc += f1[i] * fw2[i * 50 + tid];
    acc = (acc - rm2[tid]) * (g2[tid] * rsqrtf(rv2[tid] + 1e-5f)) + be2[tid];
    f2[tid] = acc > 0.0f ? acc : 0.0f;
  }
  __syncthreads();
  if (tid < 50) {
    float acc = fb3[tid];
    for (int i = 0; i < 50; ++i) acc += f2[i] * fw3[i * 50 + tid];
    lat[tid] = acc;
    latent_ws[tid] = acc;
    out_latent[tid] = acc;
  }
  __syncthreads();
  if (tid < 20) {
    float acc = db1[tid];
    for (int i = 0; i < 50; ++i) acc += lat[i] * dw1[(6 + i) * 20 + tid];
    bias1_eff[tid] = acc;
  }
  if (tid < 3) {
    float acc = dec_bias[tid];
    for (int i = 0; i < 50; ++i) acc += lat[i] * dec_root[i * 3 + tid];
    c3[tid] = acc;
  }
}

// ---------------- Decoder: per-edge MLP (6(+folded latent)->20->20->20->150) + msg + scatter ----------------
__global__ __launch_bounds__(TPB) void dec_edge_kernel(
    const int* __restrict__ ei, const float* __restrict__ ea,
    const float* __restrict__ dw1, const float* __restrict__ bias1_eff,
    const float* __restrict__ dw2, const float* __restrict__ db2,
    const float* __restrict__ dw3, const float* __restrict__ db3,
    const float* __restrict__ dw4, const float* __restrict__ db4,
    const float* __restrict__ latent, float* __restrict__ s_dec, int N, int E)
{
  __shared__ _Float16 sh_in [WAVES][16 * 32];
  __shared__ _Float16 sh_act[WAVES][16 * 32];
  __shared__ float    sh_out[WAVES][16 * 160];
  __shared__ int      sh_dst[WAVES][16];
  __shared__ float    sh_lat[50];

  if (threadIdx.x < 50) sh_lat[threadIdx.x] = latent[threadIdx.x];
  __syncthreads();

  const int lane = threadIdx.x & 31;
  const int wid  = threadIdx.x >> 5;
  _Float16* in16 = sh_in[wid];
  _Float16* act  = sh_act[wid];
  float*    wout = sh_out[wid];
  int*      dsti = sh_dst[wid];

  const int n15   = lane & 15;
  const int mbase = (lane & 16) ? 8 : 0;

  {
    unsigned* z = (unsigned*)in16;
    #pragma unroll
    for (int i = 0; i < 8; ++i) z[lane + 32 * i] = 0u;
  }

  v16h b1[2], b2[2], b3[2], b4[10];
  float bi1[2], bi2[2], bi3[2], bi4[10];
  #pragma unroll
  for (int t = 0; t < 2; ++t) {
    b1[t] = load_bfrag(dw1, 6, 20, t, lane);   // rows 0..5; latent part folded into bias1_eff
    b2[t] = load_bfrag(dw2, 20, 20, t, lane);
    b3[t] = load_bfrag(dw3, 20, 20, t, lane);
    int col = t * 16 + n15;
    bi1[t] = (col < 20) ? bias1_eff[col] : 0.0f;
    bi2[t] = (col < 20) ? db2[col] : 0.0f;
    bi3[t] = (col < 20) ? db3[col] : 0.0f;
  }
  #pragma unroll
  for (int t = 0; t < 10; ++t) {
    b4[t] = load_bfrag(dw4, 20, 150, t, lane);
    int col = t * 16 + n15;
    bi4[t] = (col < 150) ? db4[col] : 0.0f;
  }
  WAIT_DS();

  const int nTiles = (E + 15) >> 4;
  const int gw = blockIdx.x * WAVES + wid;
  const int nw = gridDim.x * WAVES;

  for (int tile = gw; tile < nTiles; tile += nw) {
    const int e0 = tile << 4;

    // Stage edge_attr[:, :6] tile into LDS f16 (3 coalesced loads per lane).
    {
      int row  = lane >> 1;
      int colb = (lane & 1) * 3;
      float a0 = 0.f, a1 = 0.f, a2 = 0.f;
      if (e0 + row < E) {
        const float* p = ea + (size_t)(e0 + row) * 12 + colb;
        a0 = p[0]; a1 = p[1]; a2 = p[2];
      }
      _Float16* q = in16 + row * 32 + colb;
      q[0] = (_Float16)a0; q[1] = (_Float16)a1; q[2] = (_Float16)a2;
    }
    WAIT_DS();
    v16h a = load_afrag_lds(in16, lane);

    // Layer 1
    {
      v8f c0 = {}, c1 = {};
      c0 = WMMA_F16(a, b1[0], c0);
      c1 = WMMA_F16(a, b1[1], c1);
      #pragma unroll
      for (int r = 0; r < 8; ++r) {
        float v0 = c0[r] + bi1[0];
        float v1 = c1[r] + bi1[1];
        act[(mbase + r) * 32 + n15]      = (_Float16)(v0 > 0.0f ? v0 : 0.0f);
        act[(mbase + r) * 32 + 16 + n15] = (_Float16)(v1 > 0.0f ? v1 : 0.0f);
      }
    }
    WAIT_DS();
    a = load_afrag_lds(act, lane);

    // Layer 2
    {
      v8f c0 = {}, c1 = {};
      c0 = WMMA_F16(a, b2[0], c0);
      c1 = WMMA_F16(a, b2[1], c1);
      #pragma unroll
      for (int r = 0; r < 8; ++r) {
        float v0 = c0[r] + bi2[0];
        float v1 = c1[r] + bi2[1];
        act[(mbase + r) * 32 + n15]      = (_Float16)(v0 > 0.0f ? v0 : 0.0f);
        act[(mbase + r) * 32 + 16 + n15] = (_Float16)(v1 > 0.0f ? v1 : 0.0f);
      }
    }
    WAIT_DS();
    a = load_afrag_lds(act, lane);

    // Layer 3
    {
      v8f c0 = {}, c1 = {};
      c0 = WMMA_F16(a, b3[0], c0);
      c1 = WMMA_F16(a, b3[1], c1);
      #pragma unroll
      for (int r = 0; r < 8; ++r) {
        float v0 = c0[r] + bi3[0];
        float v1 = c1[r] + bi3[1];
        act[(mbase + r) * 32 + n15]      = (_Float16)(v0 > 0.0f ? v0 : 0.0f);
        act[(mbase + r) * 32 + 16 + n15] = (_Float16)(v1 > 0.0f ? v1 : 0.0f);
      }
    }
    WAIT_DS();
    a = load_afrag_lds(act, lane);

    // Layer 4
    #pragma unroll
    for (int t = 0; t < 10; t += 2) {
      v8f c0 = {}, c1 = {};
      c0 = WMMA_F16(a, b4[t], c0);
      c1 = WMMA_F16(a, b4[t + 1], c1);
      #pragma unroll
      for (int r = 0; r < 8; ++r) {
        wout[(mbase + r) * 160 + t * 16 + n15]      = c0[r] + bi4[t];
        wout[(mbase + r) * 160 + t * 16 + 16 + n15] = c1[r] + bi4[t + 1];
      }
    }

    if (lane < 16) {
      int ee = e0 + lane;
      dsti[lane] = (ee < E) ? ei[E + ee] : -1;
    }
    WAIT_DS();

    // msg[e][j] = sum_i latent[i] * W[e][i*3+j]; scatter-add into s_dec[dst][j]
    #pragma unroll
    for (int t = 0; t < 2; ++t) {
      int idx = lane + 32 * t;  // 0..63, use 0..47
      if (idx < 48) {
        int el = idx / 3;
        int j  = idx - el * 3;
        int d  = dsti[el];
        if (d >= 0) {
          const float* wr = wout + el * 160;
          float acc = 0.0f;
          for (int i = 0; i < 50; ++i) acc += sh_lat[i] * wr[i * 3 + j];
          atomicAdd(&s_dec[d * 3 + j], acc);
        }
      }
    }
  }
}

// ---------------- Finalize out ----------------
__global__ __launch_bounds__(256) void out_kernel(
    const float* __restrict__ s_dec, const float* __restrict__ cnt,
    const float* __restrict__ c3, float* __restrict__ out, int N)
{
  int idx = blockIdx.x * blockDim.x + threadIdx.x;
  if (idx < N * 3) {
    int n = idx / 3;
    int j = idx - n * 3;
    float c = cnt[n]; c = c < 1.0f ? 1.0f : c;
    out[idx] = s_dec[idx] / c + c3[j];
  }
}

extern "C" void kernel_launch(void* const* d_in, const int* in_sizes, int n_in,
                              void* d_out, int out_size, void* d_ws, size_t ws_size,
                              hipStream_t stream) {
  const float* x        = (const float*)d_in[0];
  const int*   ei       = (const int*)  d_in[1];
  const float* ea       = (const float*)d_in[2];
  const float* ew1      = (const float*)d_in[3];
  const float* eb1      = (const float*)d_in[4];
  const float* ew2      = (const float*)d_in[5];
  const float* eb2      = (const float*)d_in[6];
  const float* ew3      = (const float*)d_in[7];
  const float* eb3      = (const float*)d_in[8];
  const float* ew4      = (const float*)d_in[9];
  const float* eb4      = (const float*)d_in[10];
  const float* enc_root = (const float*)d_in[11];
  const float* enc_bias = (const float*)d_in[12];
  const float* fw1      = (const float*)d_in[13];
  const float* fb1      = (const float*)d_in[14];
  const float* g1       = (const float*)d_in[15];
  const float* be1      = (const float*)d_in[16];
  const float* rm1      = (const float*)d_in[17];
  const float* rv1      = (const float*)d_in[18];
  const float* fw2      = (const float*)d_in[19];
  const float* fb2      = (const float*)d_in[20];
  const float* g2       = (const float*)d_in[21];
  const float* be2      = (const float*)d_in[22];
  const float* rm2      = (const float*)d_in[23];
  const float* rv2      = (const float*)d_in[24];
  const float* fw3      = (const float*)d_in[25];
  const float* fb3      = (const float*)d_in[26];
  const float* dw1      = (const float*)d_in[27];
  const float* db1      = (const float*)d_in[28];
  const float* dw2      = (const float*)d_in[29];
  const float* db2      = (const float*)d_in[30];
  const float* dw3      = (const float*)d_in[31];
  const float* db3      = (const float*)d_in[32];
  const float* dw4      = (const float*)d_in[33];
  const float* db4      = (const float*)d_in[34];
  const float* dec_root = (const float*)d_in[35];
  const float* dec_bias = (const float*)d_in[36];

  const int N = in_sizes[0] / 3;
  const int E = in_sizes[1] / 2;

  float* ws        = (float*)d_ws;
  float* s_enc     = ws;                          // N*50
  float* cnt       = s_enc + (size_t)N * 50;      // N
  float* pooled    = cnt + N;                     // 50
  float* s_dec     = pooled + 50;                 // N*3
  float* latent    = s_dec + (size_t)N * 3;       // 50
  float* bias1_eff = latent + 50;                 // 20
  float* c3        = bias1_eff + 20;              // 3

  const size_t zero_floats = (size_t)N * 50 + N + 50 + (size_t)N * 3;
  hipMemsetAsync(d_ws, 0, zero_floats * sizeof(float), stream);

  float* outp = (float*)d_out;

  enc_edge_kernel<<<1024, TPB, 0, stream>>>(x, ei, ea, ew1, eb1, ew2, eb2, ew3, eb3,
                                            ew4, eb4, s_enc, cnt, N, E);

  int hp_blocks = (N * 50 + 255) / 256;
  h_pool_kernel<<<hp_blocks, 256, 0, stream>>>(s_enc, cnt, x, enc_root, enc_bias, pooled, N);

  latent_kernel<<<1, 64, 0, stream>>>(pooled, N, fw1, fb1, g1, be1, rm1, rv1,
                                      fw2, fb2, g2, be2, rm2, rv2, fw3, fb3,
                                      dw1, db1, dec_root, dec_bias,
                                      latent, bias1_eff, c3, outp + (size_t)N * 3);

  dec_edge_kernel<<<1024, TPB, 0, stream>>>(ei, ea, dw1, bias1_eff, dw2, db2,
                                            dw3, db3, dw4, db4, latent, s_dec, N, E);

  out_kernel<<<(N * 3 + 255) / 256, 256, 0, stream>>>(s_dec, cnt, c3, outp, N);
}
